// Seq2Seq_47991964565597
// MI455X (gfx1250) — matrix operand order repaired and checked by reading, not compiled, not run on hardware
//
#include <hip/hip_runtime.h>
#include <hip/hip_bf16.h>
#include <math.h>

typedef __attribute__((ext_vector_type(16))) _Float16 v16h;
typedef __attribute__((ext_vector_type(8)))  _Float16 v8h;
typedef __attribute__((ext_vector_type(8)))  float    v8f;

union V16H { v16h v; v8h h[2]; };

__device__ __forceinline__ v8f wmma_f16(v16h a, v16h b, v8f c) {
    return __builtin_amdgcn_wmma_f32_16x16x32_f16(false, a, false, b, (short)0, c, false, false);
}
__device__ __forceinline__ float sigm(float x) { return 1.0f / (1.0f + expf(-x)); }

// Async DMA one 16-byte chunk global -> LDS (ASYNCcnt-tracked, no VGPR transit).
// LDS operand is the byte offset (low 32 bits of the generic shared pointer).
__device__ __forceinline__ void async_cp16(void* lds, const void* gmem) {
    asm volatile("global_load_async_to_lds_b128 %0, %1, off"
                 :: "v"((unsigned)(size_t)lds), "v"(gmem) : "memory");
}
__device__ __forceinline__ void wait_async0() {
    asm volatile("s_wait_asynccnt 0x0" ::: "memory");
}

// ---------------------------------------------------------------------------
// Generic f16 GEMM:  C[M,N] = A[M,K] @ W[N,K]^T (+bias)
// A row-major f16, W row-major f16 (torch Linear weight), C f32.
// The 16xK A tile of a block (contiguous rows) is staged into LDS once via
// GLOBAL_LOAD_ASYNC_TO_LDS_B128; all 8 waves then read A-fragments from LDS
// while B streams from global (prefetched).  One 16x16 WMMA tile per wave.
// row_mode: 0 identity, 1 ctc map (r=t*16+b -> b*512+t), 2 dec map (-> b*512+t+1)
// ---------------------------------------------------------------------------
__global__ void __launch_bounds__(256)
gemm_nt_kernel(const _Float16* __restrict__ A, const _Float16* __restrict__ W,
               const float* __restrict__ bias, float* __restrict__ C,
               int M, int N, int K, int ldc, int row_mode)
{
    extern __shared__ char smraw[];
    _Float16* smA = (_Float16*)smraw;                  // 16 x K halves (<= 16 KB)

    const int tid  = threadIdx.x;
    const int lane = tid & 31;
    const int wave = tid >> 5;
    const int m0 = blockIdx.y * 16;
    const int n0 = blockIdx.x * 128 + wave * 16;

    // ---- stage A tile (rows m0..m0+15 are one contiguous 32K-byte span) ----
    {
        const char* gsrc = (const char*)(A + (size_t)m0 * K);
        const int chunks = (16 * K * 2) >> 4;          // #16B chunks = 2K
        for (int i = tid; i < chunks; i += 256)
            async_cp16(smraw + ((size_t)i << 4), gsrc + ((size_t)i << 4));
        wait_async0();
    }
    __syncthreads();

    if (n0 < N) {                                      // wave-uniform branch
        const int mrow = lane & 15;
        const int kbA  = (lane >> 4) << 3;             // 0 / 8  (halves)
        const int nrow = lane & 15;
        const int kbB  = (lane >> 4) << 3;             // 0 / 8  (halves)

        const _Float16* Ap = smA + mrow * K + kbA;     // LDS
        const _Float16* Wp = W + (size_t)(n0 + nrow) * K + kbB;

        v8f acc = {};
        for (int k0 = 0; k0 < K; k0 += 32) {
            __builtin_prefetch(Wp + k0 + 256, 0, 0);   // stream weights
            V16H af, bf;
            af.h[0] = *reinterpret_cast<const v8h*>(Ap + k0);
            af.h[1] = *reinterpret_cast<const v8h*>(Ap + k0 + 16);
            bf.h[0] = *reinterpret_cast<const v8h*>(Wp + k0);
            bf.h[1] = *reinterpret_cast<const v8h*>(Wp + k0 + 16);
            acc = wmma_f16(af.v, bf.v, acc);
        }
        const int col   = lane & 15;
        const int rbase = (lane >> 4) << 3;
        const float bn  = bias ? bias[n0 + col] : 0.0f;
        for (int r = 0; r < 8; ++r) {
            int rg = m0 + rbase + r;
            int orow;
            if (row_mode == 0) orow = rg;
            else { int b = rg & 15, t = rg >> 4; orow = b * 512 + t + (row_mode == 2 ? 1 : 0); }
            C[(size_t)orow * ldc + n0 + col] = acc[r] + bn;
        }
    }
}

// ---------------------------------------------------------------------------
// Persistent encoder LSTM recurrence (one direction, one layer, one WG).
// Gx[t*16+b, 1024] = precomputed x@Wih^T.  State h (f16) + c (f32) in LDS.
// Per step: gates_rec = h @ Whh^T via WMMA (M=16,N=1024,K=256), then
// elementwise LSTM update.  yout[t*16+b, col0 .. col0+255] <- h_t (f16).
// ---------------------------------------------------------------------------
__global__ void __launch_bounds__(256)
lstm_enc_kernel(const float* __restrict__ Gx, const _Float16* __restrict__ Whh,
                const float* __restrict__ bih, const float* __restrict__ bhh,
                _Float16* __restrict__ yout, int col0,
                float* __restrict__ hfin, float* __restrict__ cfin,
                int T, int rev)
{
    extern __shared__ char sm[];
    _Float16* h16 = (_Float16*)sm;                          // 16*256 f16 ( 8 KB)
    float*    cst = (float*)(sm + 16*256*2);                // 16*256 f32 (16 KB)
    float*    gsm = (float*)(sm + 16*256*2 + 16*256*4);     // 16*1024 f32 (64 KB)

    const int tid = threadIdx.x, lane = tid & 31, wave = tid >> 5;
    for (int i = tid; i < 16*256; i += 256) { h16[i] = (_Float16)0.0f; cst[i] = 0.0f; }
    __syncthreads();

    const int mrow = lane & 15, kbA = (lane >> 4) << 3;
    const int nrow = lane & 15, kbB = (lane >> 4) << 3;
    const int col  = lane & 15, rb  = (lane >> 4) << 3;

    for (int s = 0; s < T; ++s) {
        const int t = rev ? (T - 1 - s) : s;
        // --- recurrent GEMM: 8 waves x 8 tiles cover N=1024 ---
        for (int tt = 0; tt < 8; ++tt) {
            const int n0 = (wave * 8 + tt) * 16;
            const _Float16* hp = h16 + mrow * 256 + kbA;
            const _Float16* wp = Whh + (size_t)(n0 + nrow) * 256 + kbB;
            v8f acc = {};
            for (int k0 = 0; k0 < 256; k0 += 32) {
                V16H af, bf;
                af.h[0] = *reinterpret_cast<const v8h*>(hp + k0);
                af.h[1] = *reinterpret_cast<const v8h*>(hp + k0 + 16);
                bf.h[0] = *reinterpret_cast<const v8h*>(wp + k0);
                bf.h[1] = *reinterpret_cast<const v8h*>(wp + k0 + 16);
                acc = wmma_f16(af.v, bf.v, acc);
            }
            for (int r = 0; r < 8; ++r) gsm[(rb + r) * 1024 + n0 + col] = acc[r];
        }
        __syncthreads();
        // --- elementwise LSTM cell (torch gate order i,f,g,o) ---
        for (int kk = 0; kk < 16; ++kk) {
            const int idx = tid + kk * 256; const int b = idx >> 8, j = idx & 255;
            const float* gx = Gx + ((size_t)t * 16 + b) * 1024;
            const float gi = gsm[b*1024 +        j] + gx[       j] + bih[       j] + bhh[       j];
            const float gf = gsm[b*1024 +  256 + j] + gx[ 256 + j] + bih[ 256 + j] + bhh[ 256 + j];
            const float gg = gsm[b*1024 +  512 + j] + gx[ 512 + j] + bih[ 512 + j] + bhh[ 512 + j];
            const float go = gsm[b*1024 +  768 + j] + gx[ 768 + j] + bih[ 768 + j] + bhh[ 768 + j];
            const float cn = sigm(gf) * cst[b*256 + j] + sigm(gi) * tanhf(gg);
            const float hn = sigm(go) * tanhf(cn);
            cst[b*256 + j] = cn;
            h16[b*256 + j] = (_Float16)hn;
            yout[((size_t)t * 16 + b) * 512 + col0 + j] = (_Float16)hn;
        }
        __syncthreads();
    }
    for (int i = tid; i < 16*256; i += 256) { hfin[i] = (float)h16[i]; cfin[i] = cst[i]; }
}

// ---------------------------------------------------------------------------
// Persistent attention decoder: single WG, 1024 threads (32 waves), state in
// LDS.  Per step: q = h@W2^T (WMMA) ; additive attention + softmax + context;
// relu(dec_emb) ; gates = [ctx|e]@Wih_d^T + h@Whh_d^T (WMMA, K=768 then 512);
// LSTM update; h_t recorded to hsdec for the batched Wdec GEMM afterwards.
// ---------------------------------------------------------------------------
__global__ void __launch_bounds__(1024)
decoder_kernel(const float* __restrict__ encproj, const _Float16* __restrict__ enc16,
               const _Float16* __restrict__ W2h, const float* __restrict__ b2,
               const float* __restrict__ Vw, const float* __restrict__ bV,
               const _Float16* __restrict__ demb, const int* __restrict__ yids,
               const _Float16* __restrict__ Wihd, const _Float16* __restrict__ Whhd,
               const float* __restrict__ bihd, const float* __restrict__ bhhd,
               const float* __restrict__ h0, const float* __restrict__ c0,
               _Float16* __restrict__ hsdec, int TDEC)
{
    extern __shared__ char sm[];
    _Float16* h16 = (_Float16*)sm;                         //  16*512 f16  (16 KB)
    _Float16* inb = (_Float16*)(sm + 16384);               //  16*768 f16  (24 KB) ctx|e
    float*    cst = (float*)(sm + 40960);                  //  16*512 f32  (32 KB)
    float*    qv  = (float*)(sm + 73728);                  //  16*128 f32  ( 8 KB)
    float*    sc  = (float*)(sm + 81920);                  //  16*512 f32  (32 KB)
    float*    gsm = (float*)(sm + 114688);                 //  16*2048 f32 (128 KB)

    const int tid = threadIdx.x, lane = tid & 31, wave = tid >> 5;
    const int mrow = lane & 15, kbA = (lane >> 4) << 3;
    const int nrow = lane & 15, kbB = (lane >> 4) << 3;
    const int col  = lane & 15, rb  = (lane >> 4) << 3;

    for (int i = tid; i < 16*512; i += 1024) { h16[i] = (_Float16)h0[i]; cst[i] = c0[i]; }
    __syncthreads();

    for (int t = 0; t < TDEC; ++t) {
        // ---- q = h @ W2^T + b2 : [16,128], waves 0..7 ----
        if (wave < 8) {
            const int n0 = wave * 16;
            const _Float16* hp = h16 + mrow * 512 + kbA;
            const _Float16* wp = W2h + (size_t)(n0 + nrow) * 512 + kbB;
            v8f acc = {};
            for (int k0 = 0; k0 < 512; k0 += 32) {
                V16H af, bf;
                af.h[0] = *reinterpret_cast<const v8h*>(hp + k0);
                af.h[1] = *reinterpret_cast<const v8h*>(hp + k0 + 16);
                bf.h[0] = *reinterpret_cast<const v8h*>(wp + k0);
                bf.h[1] = *reinterpret_cast<const v8h*>(wp + k0 + 16);
                acc = wmma_f16(af.v, bf.v, acc);
            }
            for (int r = 0; r < 8; ++r) qv[(rb + r) * 128 + n0 + col] = acc[r] + b2[n0 + col];
        }
        __syncthreads();
        // ---- scores: sc[b,s] = tanh(enc_proj[b,s]+q[b]) . V + bV ----
        for (int kk = 0; kk < 8; ++kk) {
            const int idx = tid + kk * 1024; const int b = idx >> 9, s = idx & 511;
            const float* ep = encproj + ((size_t)s * 16 + b) * 128;
            float a = bV[0];
            for (int ai = 0; ai < 128; ++ai) a += tanhf(ep[ai] + qv[b*128 + ai]) * Vw[ai];
            sc[b*512 + s] = a;
        }
        __syncthreads();
        // ---- softmax over S=512 : one wave per batch row ----
        if (wave < 16) {
            const int b = wave;
            float mx = -3.4e38f;
            for (int s = lane; s < 512; s += 32) mx = fmaxf(mx, sc[b*512 + s]);
            for (int m = 16; m; m >>= 1) mx = fmaxf(mx, __shfl_xor(mx, m, 32));
            float sum = 0.0f;
            for (int s = lane; s < 512; s += 32) sum += expf(sc[b*512 + s] - mx);
            for (int m = 16; m; m >>= 1) sum += __shfl_xor(sum, m, 32);
            const float inv = 1.0f / sum;
            for (int s = lane; s < 512; s += 32) sc[b*512 + s] = expf(sc[b*512 + s] - mx) * inv;
        }
        __syncthreads();
        // ---- ctx[b] = sum_s w[s]*enc_out[b,s]  (2 waves per batch row) ----
        {
            const int b = wave >> 1, half = wave & 1;
            const int d0 = half * 256 + lane * 8;
            float a8[8] = {0,0,0,0,0,0,0,0};
            for (int s = 0; s < 512; ++s) {
                const float wt = sc[b*512 + s];
                const v8h ev = *reinterpret_cast<const v8h*>(enc16 + ((size_t)s * 16 + b) * 512 + d0);
                for (int j = 0; j < 8; ++j) a8[j] += wt * (float)ev[j];
            }
            for (int j = 0; j < 8; ++j) inb[b*768 + d0 + j] = (_Float16)a8[j];
        }
        // ---- e = relu(dec_emb[tok]) into inb cols 512..767 ----
        for (int kk = 0; kk < 4; ++kk) {
            const int idx = tid + kk * 1024; const int b = idx >> 8, j = idx & 255;
            const int tok = yids[b * 512 + t];
            float v = (float)demb[(size_t)tok * 256 + j];
            inb[b*768 + 512 + j] = (_Float16)(v > 0.0f ? v : 0.0f);
        }
        __syncthreads();
        // ---- gates = [ctx|e]@Wih_d^T + h@Whh_d^T : [16,2048], 32 waves x 4 tiles ----
        for (int tt = 0; tt < 4; ++tt) {
            const int n0 = (wave * 4 + tt) * 16;
            v8f acc = {};
            const _Float16* ip = inb + mrow * 768 + kbA;
            const _Float16* wp = Wihd + (size_t)(n0 + nrow) * 768 + kbB;
            for (int k0 = 0; k0 < 768; k0 += 32) {
                V16H af, bf;
                af.h[0] = *reinterpret_cast<const v8h*>(ip + k0);
                af.h[1] = *reinterpret_cast<const v8h*>(ip + k0 + 16);
                bf.h[0] = *reinterpret_cast<const v8h*>(wp + k0);
                bf.h[1] = *reinterpret_cast<const v8h*>(wp + k0 + 16);
                acc = wmma_f16(af.v, bf.v, acc);
            }
            const _Float16* hp = h16 + mrow * 512 + kbA;
            const _Float16* vp = Whhd + (size_t)(n0 + nrow) * 512 + kbB;
            for (int k0 = 0; k0 < 512; k0 += 32) {
                V16H af, bf;
                af.h[0] = *reinterpret_cast<const v8h*>(hp + k0);
                af.h[1] = *reinterpret_cast<const v8h*>(hp + k0 + 16);
                bf.h[0] = *reinterpret_cast<const v8h*>(vp + k0);
                bf.h[1] = *reinterpret_cast<const v8h*>(vp + k0 + 16);
                acc = wmma_f16(af.v, bf.v, acc);
            }
            for (int r = 0; r < 8; ++r) gsm[(rb + r) * 2048 + n0 + col] = acc[r];
        }
        __syncthreads();
        // ---- LSTM cell update, record h_t ----
        for (int kk = 0; kk < 8; ++kk) {
            const int idx = tid + kk * 1024; const int b = idx >> 9, j = idx & 511;
            const float gi = gsm[b*2048 +         j] + bihd[        j] + bhhd[        j];
            const float gf = gsm[b*2048 +  512 +  j] + bihd[ 512 +  j] + bhhd[ 512 +  j];
            const float gg = gsm[b*2048 + 1024 +  j] + bihd[1024 +  j] + bhhd[1024 +  j];
            const float go = gsm[b*2048 + 1536 +  j] + bihd[1536 +  j] + bhhd[1536 +  j];
            const float cn = sigm(gf) * cst[b*512 + j] + sigm(gi) * tanhf(gg);
            const float hn = sigm(go) * tanhf(cn);
            cst[b*512 + j] = cn;
            h16[b*512 + j] = (_Float16)hn;
            hsdec[((size_t)t * 16 + b) * 512 + j] = (_Float16)hn;
        }
        __syncthreads();
    }
}

// ---------------------------------------------------------------------------
// Small helper kernels
// ---------------------------------------------------------------------------
__global__ void cvt_kernel(const float* __restrict__ src, _Float16* __restrict__ dst, int n) {
    int idx = blockIdx.x * 256 + threadIdx.x;
    if (idx < n) dst[idx] = (_Float16)src[idx];
}

__global__ void embed_kernel(const int* __restrict__ x, const float* __restrict__ emb,
                             _Float16* __restrict__ out) {
    int idx = blockIdx.x * 256 + threadIdx.x;          // 8192*256
    if (idx >= 8192 * 256) return;
    int row = idx >> 8, j = idx & 255;
    int t = row >> 4, b = row & 15;
    int tok = x[b * 512 + t];
    out[idx] = (_Float16)emb[(size_t)tok * 256 + j];
}

__global__ void h0c0_kernel(const float* hf0f, const float* hf0b, const float* hf1f, const float* hf1b,
                            const float* cf0f, const float* cf0b, const float* cf1f, const float* cf1b,
                            float* h0, float* c0) {
    int idx = blockIdx.x * 256 + threadIdx.x;
    if (idx >= 16 * 512) return;
    int b = idx >> 9, j = idx & 511, jj = j & 255;
    const float* ha = (j < 256) ? hf0f : hf0b; const float* hb = (j < 256) ? hf1f : hf1b;
    const float* ca = (j < 256) ? cf0f : cf0b; const float* cb = (j < 256) ? cf1f : cf1b;
    h0[idx] = 0.5f * (ha[b*256 + jj] + hb[b*256 + jj]);
    c0[idx] = 0.5f * (ca[b*256 + jj] + cb[b*256 + jj]);
}

__global__ void __launch_bounds__(256)
logsoftmax_kernel(float* __restrict__ X, int rows, int n) {
    const int wave = threadIdx.x >> 5, lane = threadIdx.x & 31;
    const int row = blockIdx.x * 8 + wave;
    if (row >= rows) return;
    float* p = X + (size_t)row * n;
    float mx = -3.4e38f;
    for (int i = lane; i < n; i += 32) mx = fmaxf(mx, p[i]);
    for (int m = 16; m; m >>= 1) mx = fmaxf(mx, __shfl_xor(mx, m, 32));
    float sum = 0.0f;
    for (int i = lane; i < n; i += 32) sum += expf(p[i] - mx);
    for (int m = 16; m; m >>= 1) sum += __shfl_xor(sum, m, 32);
    const float lse = mx + logf(sum);
    for (int i = lane; i < n; i += 32) p[i] -= lse;
}

__global__ void zero_t0_kernel(float* __restrict__ out2) {
    int idx = blockIdx.x * 256 + threadIdx.x;
    if (idx >= 16 * 8000) return;
    int b = idx / 8000, nn = idx % 8000;
    out2[(size_t)b * 512 * 8000 + nn] = 0.0f;
}

// ---------------------------------------------------------------------------
extern "C" void kernel_launch(void* const* d_in, const int* in_sizes, int n_in,
                              void* d_out, int out_size, void* d_ws, size_t ws_size,
                              hipStream_t stream) {
    (void)in_sizes; (void)n_in; (void)out_size; (void)ws_size;
    const int*   x       = (const int*)  d_in[0];
    const int*   y       = (const int*)  d_in[1];
    const float* enc_emb = (const float*)d_in[2];

    char* ws = (char*)d_ws;
    size_t off = 0;
    auto alloc = [&](size_t bytes) -> void* {
        void* p = (void*)(ws + off);
        off += (bytes + 255) & ~(size_t)255;
        return p;
    };
    // f16 weight copies
    _Float16* wWih0f = (_Float16*)alloc(1024*256*2);
    _Float16* wWhh0f = (_Float16*)alloc(1024*256*2);
    _Float16* wWih0b = (_Float16*)alloc(1024*256*2);
    _Float16* wWhh0b = (_Float16*)alloc(1024*256*2);
    _Float16* wWih1f = (_Float16*)alloc(1024*512*2);
    _Float16* wWhh1f = (_Float16*)alloc(1024*256*2);
    _Float16* wWih1b = (_Float16*)alloc(1024*512*2);
    _Float16* wWhh1b = (_Float16*)alloc(1024*256*2);
    _Float16* wW1    = (_Float16*)alloc(128*512*2);
    _Float16* wW2    = (_Float16*)alloc(128*512*2);
    _Float16* wWihd  = (_Float16*)alloc(2048*768*2);
    _Float16* wWhhd  = (_Float16*)alloc(2048*512*2);
    _Float16* wWdec  = (_Float16*)alloc((size_t)8000*512*2);
    _Float16* wWctc  = (_Float16*)alloc((size_t)8000*512*2);
    _Float16* wDemb  = (_Float16*)alloc((size_t)8000*256*2);
    // activations
    _Float16* emb16  = (_Float16*)alloc((size_t)8192*256*2);
    _Float16* l0out  = (_Float16*)alloc((size_t)8192*512*2);
    _Float16* l1out  = (_Float16*)alloc((size_t)8192*512*2);
    _Float16* hsdec  = (_Float16*)alloc((size_t)8192*512*2);
    float*    Gx     = (float*)alloc((size_t)8192*1024*4);
    float*    encproj= (float*)alloc((size_t)8192*128*4);
    float*    hf0f = (float*)alloc(16*256*4); float* cf0f = (float*)alloc(16*256*4);
    float*    hf0b = (float*)alloc(16*256*4); float* cf0b = (float*)alloc(16*256*4);
    float*    hf1f = (float*)alloc(16*256*4); float* cf1f = (float*)alloc(16*256*4);
    float*    hf1b = (float*)alloc(16*256*4); float* cf1b = (float*)alloc(16*256*4);
    float*    h0   = (float*)alloc(16*512*4); float* c0   = (float*)alloc(16*512*4);

    // ---- f32 -> f16 weight conversions ----
    struct Cvt { int idx; _Float16* dst; int n; };
    const Cvt cv[] = {
        {3,  wWih0f, 1024*256}, {4,  wWhh0f, 1024*256},
        {7,  wWih0b, 1024*256}, {8,  wWhh0b, 1024*256},
        {11, wWih1f, 1024*512}, {12, wWhh1f, 1024*256},
        {15, wWih1b, 1024*512}, {16, wWhh1b, 1024*256},
        {19, wW1,    128*512},  {21, wW2,    128*512},
        {25, wDemb,  8000*256}, {26, wWihd,  2048*768},
        {27, wWhhd,  2048*512}, {30, wWdec,  8000*512},
        {32, wWctc,  8000*512},
    };
    for (const Cvt& c : cv)
        cvt_kernel<<<(c.n + 255) / 256, 256, 0, stream>>>((const float*)d_in[c.idx], c.dst, c.n);

    // ---- embedding gather (f32 -> f16, rows = t*16+b) ----
    embed_kernel<<<8192, 256, 0, stream>>>(x, enc_emb, emb16);

    const size_t lstm_sm = 16*256*2 + 16*256*4 + 16*1024*4;   // 90112 B
    dim3 blk(256);
    const size_t smA256 = 32 * 256;   // gemm LDS A-tile for K=256 (8 KB)
    const size_t smA512 = 32 * 512;   // gemm LDS A-tile for K=512 (16 KB)

    // ---- encoder layer 0 ----
    gemm_nt_kernel<<<dim3(8, 512), blk, smA256, stream>>>(emb16, wWih0f, nullptr, Gx, 8192, 1024, 256, 1024, 0);
    lstm_enc_kernel<<<1, blk, lstm_sm, stream>>>(Gx, wWhh0f, (const float*)d_in[5], (const float*)d_in[6],
                                                 l0out, 0,   hf0f, cf0f, 512, 0);
    gemm_nt_kernel<<<dim3(8, 512), blk, smA256, stream>>>(emb16, wWih0b, nullptr, Gx, 8192, 1024, 256, 1024, 0);
    lstm_enc_kernel<<<1, blk, lstm_sm, stream>>>(Gx, wWhh0b, (const float*)d_in[9], (const float*)d_in[10],
                                                 l0out, 256, hf0b, cf0b, 512, 1);
    // ---- encoder layer 1 ----
    gemm_nt_kernel<<<dim3(8, 512), blk, smA512, stream>>>(l0out, wWih1f, nullptr, Gx, 8192, 1024, 512, 1024, 0);
    lstm_enc_kernel<<<1, blk, lstm_sm, stream>>>(Gx, wWhh1f, (const float*)d_in[13], (const float*)d_in[14],
                                                 l1out, 0,   hf1f, cf1f, 512, 0);
    gemm_nt_kernel<<<dim3(8, 512), blk, smA512, stream>>>(l0out, wWih1b, nullptr, Gx, 8192, 1024, 512, 1024, 0);
    lstm_enc_kernel<<<1, blk, lstm_sm, stream>>>(Gx, wWhh1b, (const float*)d_in[17], (const float*)d_in[18],
                                                 l1out, 256, hf1b, cf1b, 512, 1);

    // ---- decoder initial state ----
    h0c0_kernel<<<32, 256, 0, stream>>>(hf0f, hf0b, hf1f, hf1b, cf0f, cf0b, cf1f, cf1b, h0, c0);

    // ---- enc_proj = enc_out @ W1^T + b1 ----
    gemm_nt_kernel<<<dim3(1, 512), blk, smA512, stream>>>(l1out, wW1, (const float*)d_in[20],
                                                          encproj, 8192, 128, 512, 128, 0);

    // ---- CTC logits straight into d_out[0 .. 65.5M) then in-place log_softmax ----
    float* out_ctc = (float*)d_out;
    float* out_dec = (float*)d_out + (size_t)16 * 512 * 8000;
    gemm_nt_kernel<<<dim3(63, 512), blk, smA512, stream>>>(l1out, wWctc, (const float*)d_in[33],
                                                           out_ctc, 8192, 8000, 512, 8000, 1);
    logsoftmax_kernel<<<1024, 256, 0, stream>>>(out_ctc, 8192, 8000);

    // ---- sequential attention decoder (persistent single WG) ----
    decoder_kernel<<<1, 1024, 245760, stream>>>(
        encproj, l1out, wW2, (const float*)d_in[22],
        (const float*)d_in[23], (const float*)d_in[24],
        wDemb, y, wWihd, wWhhd,
        (const float*)d_in[28], (const float*)d_in[29],
        h0, c0, hsdec, 511);

    // ---- batched vocab projection of all 511 h-states into d_out ----
    gemm_nt_kernel<<<dim3(63, 511), blk, smA512, stream>>>(hsdec, wWdec, (const float*)d_in[31],
                                                           out_dec, 8176, 8000, 512, 8000, 2);
    zero_t0_kernel<<<500, 256, 0, stream>>>(out_dec);
}
